// DTIConvGraph12Layer_68745246539843
// MI455X (gfx1250) — matrix operand order repaired and checked by reading, not compiled, not run on hardware
//
#include <hip/hip_runtime.h>
#include <math.h>

#define DD   256
#define NN   16384
#define EE   262144
#define K3D  768

// ---------------------------------------------------------------------------
// CDNA5 WMMA types (wave32): A/B are 16 bf16 per lane, C/D are 8 f32 per lane
// ---------------------------------------------------------------------------
typedef __attribute__((ext_vector_type(16))) __bf16 v16bf;
typedef __attribute__((ext_vector_type(8)))  float  v8f;

union BF16Frag {
    v16bf v;
    uint4 q[2];
};

static __device__ __forceinline__ v8f wmma_bf16(const BF16Frag& a, const BF16Frag& b, v8f c) {
    // D = A(16x32 bf16) * B(32x16 bf16) + C(16x16 f32)
    return __builtin_amdgcn_wmma_f32_16x16x32_bf16(
        /*neg_a=*/false, a.v, /*neg_b=*/false, b.v,
        /*c_mod=*/(short)0, c, /*reuse_a=*/false, /*reuse_b=*/false);
}

// f32 -> bf16 (round-to-nearest-even), bit-level to avoid scalar __bf16 codegen risk
static __device__ __forceinline__ unsigned short f2bf(float f) {
    unsigned u = __float_as_uint(f);
    return (unsigned short)((u + 0x7FFFu + ((u >> 16) & 1u)) >> 16);
}

static __device__ __forceinline__ float lrelu1(float x) { return x > 0.f ? x : 0.01f   * x; }
static __device__ __forceinline__ float lrelu2(float x) { return x > 0.f ? x : 1.0e-4f * x; } // lrelu∘lrelu

// ---------------------------------------------------------------------------
// Weight conversion f32 -> bf16 bits
// ---------------------------------------------------------------------------
__global__ void k_cvt_bf16(const float* __restrict__ in, unsigned short* __restrict__ out, int n) {
    int i = blockIdx.x * 256 + threadIdx.x;
    if (i < n) out[i] = f2bf(in[i]);
}

// Re-init scratch every call (harness poisons ws): mx=-inf, denom=0, m=0
__global__ void k_init(float* __restrict__ mx, float* __restrict__ denom, float* __restrict__ m) {
    size_t i = (size_t)blockIdx.x * 256 + threadIdx.x;   // grid covers N*D
    m[i] = 0.f;
    if (i < NN) { mx[i] = -INFINITY; denom[i] = 0.f; }
}

// ---------------------------------------------------------------------------
// e = lrelu(lrelu([nf[src] | nf[dst] | ef] @ W1^T))      [E, D]
// one block = 16 edges; 8 waves each own two 16-col tiles; K = 768
// ---------------------------------------------------------------------------
__global__ __launch_bounds__(256)
void k_edge_gemm(const float* __restrict__ nf, const float* __restrict__ ef,
                 const int* __restrict__ src, const int* __restrict__ dst,
                 const unsigned short* __restrict__ w1bf, float* __restrict__ eout) {
    __shared__ __align__(16) unsigned short sA[16][K3D + 16];   // bf16 bits, padded stride
    const int tid = threadIdx.x;
    const size_t etile = blockIdx.x;

    {   // gather + convert: 16 threads per row, 48 elements each (12 x float4)
        const int r = tid >> 4;
        const size_t eidx = etile * 16 + r;
        const size_t s = (size_t)src[eidx];
        const size_t d = (size_t)dst[eidx];
        const int base = (tid & 15) * 48;
#pragma unroll
        for (int v = 0; v < 12; ++v) {
            const int i = base + v * 4;
            const float* p;
            if (i < 256)       p = nf + s * DD + i;
            else if (i < 512)  p = nf + d * DD + (i - 256);
            else               p = ef + eidx * DD + (i - 512);
            float4 f = *(const float4*)p;
            sA[r][i + 0] = f2bf(f.x);
            sA[r][i + 1] = f2bf(f.y);
            sA[r][i + 2] = f2bf(f.z);
            sA[r][i + 3] = f2bf(f.w);
        }
    }
    __syncthreads();

    const int lane = tid & 31;
    const int wave = tid >> 5;
    const int mrow = lane & 15;
    const int kha  = (lane >> 4) * 16;   // A: 16 consecutive K at kha
    const int khb  = (lane >> 4) * 16;   // B: 16 consecutive K at khb
    const int col0 = (wave)     * 16 + (lane & 15);
    const int col1 = (wave + 8) * 16 + (lane & 15);

    v8f acc0 = {}; v8f acc1 = {};
#pragma unroll 4
    for (int k0 = 0; k0 < K3D; k0 += 32) {
        BF16Frag a, b0, b1;
        a.q[0] = *(const uint4*)&sA[mrow][k0 + kha];
        a.q[1] = *(const uint4*)&sA[mrow][k0 + kha + 8];
        const unsigned short* pb0 = w1bf + (size_t)col0 * K3D + k0 + khb;
        const unsigned short* pb1 = w1bf + (size_t)col1 * K3D + k0 + khb;
        b0.q[0] = *(const uint4*)pb0; b0.q[1] = *(const uint4*)(pb0 + 8);
        b1.q[0] = *(const uint4*)pb1; b1.q[1] = *(const uint4*)(pb1 + 8);
        acc0 = wmma_bf16(a, b0, acc0);
        acc1 = wmma_bf16(a, b1, acc1);
    }

    const int rbase = (lane >> 4) * 8;   // C layout: lane>=16 holds rows M=8..15
#pragma unroll
    for (int j = 0; j < 8; ++j) {
        const size_t erow = etile * 16 + rbase + j;
        eout[erow * DD + col0] = lrelu2(acc0[j]);
        eout[erow * DD + col1] = lrelu2(acc1[j]);
    }
}

// ---------------------------------------------------------------------------
// logits + segment max: one wave per edge
// ---------------------------------------------------------------------------
__global__ __launch_bounds__(256)
void k_logits(const float* __restrict__ e, const float* __restrict__ w2,
              const int* __restrict__ dst, float* __restrict__ logit, float* __restrict__ mx) {
    const int lane = threadIdx.x & 31;
    const size_t eidx = (size_t)blockIdx.x * 8 + (threadIdx.x >> 5);
    const float* row = e + eidx * DD + lane * 8;
    const float* w   = w2 + lane * 8;
    float4 a = *(const float4*)(row);     float4 b = *(const float4*)(row + 4);
    float4 wa = *(const float4*)(w);      float4 wb = *(const float4*)(w + 4);
    float s = a.x * wa.x + a.y * wa.y + a.z * wa.z + a.w * wa.w
            + b.x * wb.x + b.y * wb.y + b.z * wb.z + b.w * wb.w;
#pragma unroll
    for (int o = 16; o > 0; o >>= 1) s += __shfl_xor(s, o, 32);
    if (lane == 0) {
        float L = lrelu2(s);
        logit[eidx] = L;
        float* addr = mx + dst[eidx];
        if (L >= 0.f) atomicMax((int*)addr, __float_as_int(L));
        else          atomicMin((unsigned int*)addr, __float_as_uint(L));
    }
}

// ex = exp(logit - mx[dst]) (in place); denom[dst] += ex
__global__ void k_exp(float* __restrict__ logit, const float* __restrict__ mx,
                      const int* __restrict__ dst, float* __restrict__ denom) {
    size_t i = (size_t)blockIdx.x * 256 + threadIdx.x;
    const int d = dst[i];
    float v = expf(logit[i] - mx[d]);
    logit[i] = v;
    atomicAdd(&denom[d], v);
}

// ---------------------------------------------------------------------------
// msg = (ex/denom[dst]) * lrelu(e @ W3^T); m[dst] += msg (scatter)
// ---------------------------------------------------------------------------
__global__ __launch_bounds__(256)
void k_msg(const float* __restrict__ e, const unsigned short* __restrict__ w3bf,
           const float* __restrict__ ex, const float* __restrict__ denom,
           const int* __restrict__ dst, float* __restrict__ m) {
    __shared__ __align__(16) unsigned short sA[16][DD + 16];
    __shared__ float sScale[16];
    __shared__ int   sDst[16];
    const int tid = threadIdx.x;
    const size_t etile = blockIdx.x;

    {
        const int r = tid >> 4;
        const size_t eidx = etile * 16 + r;
        const int base = (tid & 15) * 16;
        const float* p = e + eidx * DD + base;
#pragma unroll
        for (int v = 0; v < 4; ++v) {
            float4 f = *(const float4*)(p + v * 4);
            sA[r][base + v * 4 + 0] = f2bf(f.x);
            sA[r][base + v * 4 + 1] = f2bf(f.y);
            sA[r][base + v * 4 + 2] = f2bf(f.z);
            sA[r][base + v * 4 + 3] = f2bf(f.w);
        }
        if (tid < 16) {
            const size_t ei = etile * 16 + tid;
            const int d = dst[ei];
            sDst[tid] = d;
            sScale[tid] = ex[ei] / denom[d];
        }
    }
    __syncthreads();

    const int lane = tid & 31;
    const int wave = tid >> 5;
    const int mrow = lane & 15;
    const int kha  = (lane >> 4) * 16;
    const int khb  = (lane >> 4) * 16;
    const int col0 = (wave)     * 16 + (lane & 15);
    const int col1 = (wave + 8) * 16 + (lane & 15);

    v8f acc0 = {}; v8f acc1 = {};
#pragma unroll
    for (int k0 = 0; k0 < DD; k0 += 32) {
        BF16Frag a, b0, b1;
        a.q[0] = *(const uint4*)&sA[mrow][k0 + kha];
        a.q[1] = *(const uint4*)&sA[mrow][k0 + kha + 8];
        const unsigned short* pb0 = w3bf + (size_t)col0 * DD + k0 + khb;
        const unsigned short* pb1 = w3bf + (size_t)col1 * DD + k0 + khb;
        b0.q[0] = *(const uint4*)pb0; b0.q[1] = *(const uint4*)(pb0 + 8);
        b1.q[0] = *(const uint4*)pb1; b1.q[1] = *(const uint4*)(pb1 + 8);
        acc0 = wmma_bf16(a, b0, acc0);
        acc1 = wmma_bf16(a, b1, acc1);
    }

    const int rbase = (lane >> 4) * 8;
#pragma unroll
    for (int j = 0; j < 8; ++j) {
        const int row = rbase + j;
        const size_t dbase = (size_t)sDst[row] * DD;
        atomicAdd(&m[dbase + col0], sScale[row] * lrelu1(acc0[j]));
        atomicAdd(&m[dbase + col1], sScale[row] * lrelu1(acc1[j]));
    }
}

// ---------------------------------------------------------------------------
// GRU: gi = elu(m) @ W_ih^T + b_ih ; gh = nf @ W_hh^T + b_hh ; gates ; relu
// one block = 16 nodes; each wave owns 2 column tiles, 6 accumulators each
// ---------------------------------------------------------------------------
__global__ __launch_bounds__(256)
void k_gru(const float* __restrict__ nf, const float* __restrict__ m,
           const unsigned short* __restrict__ wih, const unsigned short* __restrict__ whh,
           const float* __restrict__ bih, const float* __restrict__ bhh,
           float* __restrict__ hout) {
    __shared__ __align__(16) unsigned short sM[16][DD + 16];
    __shared__ __align__(16) unsigned short sH[16][DD + 16];
    const int tid = threadIdx.x;
    const size_t ntile = blockIdx.x;

    {
        const int r = tid >> 4;
        const size_t node = ntile * 16 + r;
        const int base = (tid & 15) * 16;
#pragma unroll
        for (int v = 0; v < 4; ++v) {
            float4 f = *(const float4*)(m + node * DD + base + v * 4);
            f.x = f.x > 0.f ? f.x : expf(f.x) - 1.f;   // ELU
            f.y = f.y > 0.f ? f.y : expf(f.y) - 1.f;
            f.z = f.z > 0.f ? f.z : expf(f.z) - 1.f;
            f.w = f.w > 0.f ? f.w : expf(f.w) - 1.f;
            sM[r][base + v * 4 + 0] = f2bf(f.x);
            sM[r][base + v * 4 + 1] = f2bf(f.y);
            sM[r][base + v * 4 + 2] = f2bf(f.z);
            sM[r][base + v * 4 + 3] = f2bf(f.w);
            float4 g = *(const float4*)(nf + node * DD + base + v * 4);
            sH[r][base + v * 4 + 0] = f2bf(g.x);
            sH[r][base + v * 4 + 1] = f2bf(g.y);
            sH[r][base + v * 4 + 2] = f2bf(g.z);
            sH[r][base + v * 4 + 3] = f2bf(g.w);
        }
    }
    __syncthreads();

    const int lane = tid & 31;
    const int wave = tid >> 5;
    const int mrow = lane & 15;
    const int kha  = (lane >> 4) * 16;
    const int khb  = (lane >> 4) * 16;

    for (int t = 0; t < 2; ++t) {
        const int ct  = wave + t * 8;            // column tile 0..15
        const int col = ct * 16 + (lane & 15);   // output feature column
        v8f ir = {}, iz = {}, inn = {}, hr = {}, hz = {}, hn = {};
#pragma unroll
        for (int k0 = 0; k0 < DD; k0 += 32) {
            BF16Frag aM, aH, b;
            aM.q[0] = *(const uint4*)&sM[mrow][k0 + kha];
            aM.q[1] = *(const uint4*)&sM[mrow][k0 + kha + 8];
            aH.q[0] = *(const uint4*)&sH[mrow][k0 + kha];
            aH.q[1] = *(const uint4*)&sH[mrow][k0 + kha + 8];
            const unsigned short* p;
            p = wih + (size_t)(col      ) * DD + k0 + khb;
            b.q[0] = *(const uint4*)p; b.q[1] = *(const uint4*)(p + 8);
            ir  = wmma_bf16(aM, b, ir);
            p = wih + (size_t)(col + 256) * DD + k0 + khb;
            b.q[0] = *(const uint4*)p; b.q[1] = *(const uint4*)(p + 8);
            iz  = wmma_bf16(aM, b, iz);
            p = wih + (size_t)(col + 512) * DD + k0 + khb;
            b.q[0] = *(const uint4*)p; b.q[1] = *(const uint4*)(p + 8);
            inn = wmma_bf16(aM, b, inn);
            p = whh + (size_t)(col      ) * DD + k0 + khb;
            b.q[0] = *(const uint4*)p; b.q[1] = *(const uint4*)(p + 8);
            hr  = wmma_bf16(aH, b, hr);
            p = whh + (size_t)(col + 256) * DD + k0 + khb;
            b.q[0] = *(const uint4*)p; b.q[1] = *(const uint4*)(p + 8);
            hz  = wmma_bf16(aH, b, hz);
            p = whh + (size_t)(col + 512) * DD + k0 + khb;
            b.q[0] = *(const uint4*)p; b.q[1] = *(const uint4*)(p + 8);
            hn  = wmma_bf16(aH, b, hn);
        }

        const float bir = bih[col], biz = bih[col + 256], bin = bih[col + 512];
        const float bhr = bhh[col], bhz = bhh[col + 256], bhn = bhh[col + 512];
        const int rbase = (lane >> 4) * 8;
#pragma unroll
        for (int j = 0; j < 8; ++j) {
            const size_t node = ntile * 16 + rbase + j;
            const float r  = 1.f / (1.f + expf(-(ir[j] + bir + hr[j] + bhr)));
            const float z  = 1.f / (1.f + expf(-(iz[j] + biz + hz[j] + bhz)));
            const float nt = tanhf(inn[j] + bin + r * (hn[j] + bhn));
            const float hp = nf[node * DD + col];
            const float h  = (1.f - z) * nt + z * hp;
            hout[node * DD + col] = h > 0.f ? h : 0.f;
        }
    }
}

// ---------------------------------------------------------------------------
// launcher
// ---------------------------------------------------------------------------
extern "C" void kernel_launch(void* const* d_in, const int* in_sizes, int n_in,
                              void* d_out, int out_size, void* d_ws, size_t ws_size,
                              hipStream_t stream) {
    const float* nf  = (const float*)d_in[0];
    const float* ef  = (const float*)d_in[1];
    const int*   src = (const int*)  d_in[2];
    const int*   dst = (const int*)  d_in[3];
    const float* W1  = (const float*)d_in[4];
    const float* W2  = (const float*)d_in[5];
    const float* W3  = (const float*)d_in[6];
    const float* Wih = (const float*)d_in[7];
    const float* Whh = (const float*)d_in[8];
    const float* bih = (const float*)d_in[9];
    const float* bhh = (const float*)d_in[10];

    char* ws = (char*)d_ws;
    unsigned short* W1bf  = (unsigned short*)(ws + 0);        // 196608 * 2
    unsigned short* W3bf  = (unsigned short*)(ws + 393216);   //  65536 * 2
    unsigned short* Wihbf = (unsigned short*)(ws + 524288);   // 196608 * 2
    unsigned short* Whhbf = (unsigned short*)(ws + 917504);   // 196608 * 2
    float* logit = (float*)(ws + 1310720);                    // E * 4   (also ex)
    float* mx    = (float*)(ws + 2359296);                    // N * 4
    float* denom = (float*)(ws + 2424832);                    // N * 4
    float* mbuf  = (float*)(ws + 2490368);                    // N*D * 4

    float* hout = (float*)d_out;
    float* eout = (float*)d_out + (size_t)NN * DD;

    k_cvt_bf16<<<(196608 + 255) / 256, 256, 0, stream>>>(W1,  W1bf,  196608);
    k_cvt_bf16<<<( 65536 + 255) / 256, 256, 0, stream>>>(W3,  W3bf,   65536);
    k_cvt_bf16<<<(196608 + 255) / 256, 256, 0, stream>>>(Wih, Wihbf, 196608);
    k_cvt_bf16<<<(196608 + 255) / 256, 256, 0, stream>>>(Whh, Whhbf, 196608);

    k_init<<<(NN * DD) / 256, 256, 0, stream>>>(mx, denom, mbuf);

    k_edge_gemm<<<EE / 16, 256, 0, stream>>>(nf, ef, src, dst, W1bf, eout);
    k_logits  <<<EE /  8, 256, 0, stream>>>(eout, W2, dst, logit, mx);
    k_exp     <<<EE / 256, 256, 0, stream>>>(logit, mx, dst, denom);
    k_msg     <<<EE / 16, 256, 0, stream>>>(eout, W3bf, logit, denom, dst, mbuf);
    k_gru     <<<NN / 16, 256, 0, stream>>>(nf, mbuf, Wihbf, Whhbf, bih, bhh, hout);
}
